// Attention_32444182954716
// MI455X (gfx1250) — matrix operand order repaired and checked by reading, not compiled, not run on hardware
//
#include <hip/hip_runtime.h>

// ---------------------------------------------------------------------------
// Fused attention for MI455X (gfx1250, wave32, WMMA + Tensor Data Mover).
//   B=4, S=4096, D=256, fp32 in/out; f16 WMMA with f32 accumulation.
// ---------------------------------------------------------------------------

typedef __attribute__((ext_vector_type(16))) _Float16     v16h;
typedef __attribute__((ext_vector_type(8)))  float        v8f;
typedef __attribute__((ext_vector_type(4)))  unsigned int u32x4;
typedef __attribute__((ext_vector_type(8)))  int          i32x8;
typedef __attribute__((ext_vector_type(4)))  int          i32x4;
typedef __attribute__((ext_vector_type(4)))  float        f32x4;

union AFrag { v16h h; u32x4 q[2]; };

#define WMMA(A, B, C) \
  __builtin_amdgcn_wmma_f32_16x16x32_f16(false, (A), false, (B), (short)0, (C), false, false)

#if __has_builtin(__builtin_amdgcn_tensor_load_to_lds)
#define HAVE_TDM 1
#else
#define HAVE_TDM 0
#endif

constexpr int Bb = 4;
constexpr int S  = 4096;
constexpr int D  = 256;
constexpr int BS = Bb * S;
constexpr int NJ = S / 32;   // 128 key-tile steps

// ---------------------------------------------------------------------------
// Kernel 1: y = x @ W^T + b  -> f16.  One wave computes a 16x16 output tile.
//   A-fragment (16x32 f16): lane m = lane&15; half j -> k = (j<8?j:j+8)+hi*8
//   B-fragment (32x16 f16): lane n = lane&15; half j -> k = (j<8?j:j+8)+hi*8
//   C (16x16 f32): vgpr r, lane -> (row r+hi*8, col lane&15)
// transposed==1 stores V as (B, D, S) so kernel 2 gets contiguous V^T rows.
// ---------------------------------------------------------------------------
__global__ void __launch_bounds__(32)
qkv_proj_kernel(const float* __restrict__ x,
                const float* __restrict__ W,
                const float* __restrict__ bias,
                _Float16* __restrict__ dst,
                int transposed, float scale) {
  const int lane = threadIdx.x & 31;
  const int hi   = lane >> 4;
  const int nl   = lane & 15;
  const int rowTile = blockIdx.x;   // 16 tokens
  const int colTile = blockIdx.y;   // 16 output features
  const int e = colTile * 16 + nl;  // out feature for B-frag / bias
  const int m = rowTile * 16 + nl;  // token row for A-frag

  v8f acc = {};
#pragma unroll
  for (int kc = 0; kc < 8; ++kc) {
    const int d0 = kc * 32;
    AFrag a;
    const float* xr = x + (size_t)m * D + d0 + hi * 8;
    f32x4 r0 = *(const f32x4*)(xr);
    f32x4 r1 = *(const f32x4*)(xr + 4);
    f32x4 r2 = *(const f32x4*)(xr + 16);
    f32x4 r3 = *(const f32x4*)(xr + 20);
#pragma unroll
    for (int t = 0; t < 4; ++t) {
      a.h[t]      = (_Float16)r0[t];
      a.h[4 + t]  = (_Float16)r1[t];
      a.h[8 + t]  = (_Float16)r2[t];
      a.h[12 + t] = (_Float16)r3[t];
    }
    AFrag bf;
    const float* wr = W + (size_t)e * D + d0 + hi * 8;
#pragma unroll
    for (int t = 0; t < 16; t += 4) {
      f32x4 wv = *(const f32x4*)(wr + t + (t >= 8 ? 8 : 0));
      bf.h[t]     = (_Float16)wv[0];
      bf.h[t + 1] = (_Float16)wv[1];
      bf.h[t + 2] = (_Float16)wv[2];
      bf.h[t + 3] = (_Float16)wv[3];
    }
    acc = WMMA(a.h, bf.h, acc);
  }

  const float bv = bias[e];
#pragma unroll
  for (int r = 0; r < 8; ++r) {
    const float v   = (acc[r] + bv) * scale;
    const int   row = rowTile * 16 + r + hi * 8;
    if (!transposed) {
      dst[(size_t)row * D + e] = (_Float16)v;
    } else {
      const int bb = row >> 12;        // row / S
      const int ss = row & (S - 1);    // row % S
      dst[((size_t)bb * D + e) * S + ss] = (_Float16)v;
    }
  }
}

// ---------------------------------------------------------------------------
// TDM helper: 2D f16 tile load, global -> LDS, via Tensor DMA descriptor (D#).
//   D# group0: [1:0]=count=1, [63:32]=lds_addr, [120:64]=global byte addr,
//              [127:126]=type=2.
//   D# group1: [17:16]=data_size(1 -> 2B), tensor_dim0/1, tile_dim0/1,
//              tensor_dim0_stride (elements).
// Executed under a uniform one-wave guard; EXEC ignored; completion via
// TENSORcnt (s_wait_tensorcnt).
// ---------------------------------------------------------------------------
#if HAVE_TDM
__device__ __forceinline__ unsigned lds_offset(const void* p) {
  // generic LDS pointer = {SHARED_BASE, wave-relative offset[31:0]}
  return (unsigned)(unsigned long long)p;
}

__device__ __forceinline__ void tdm_load_tile_2d(unsigned lds_addr,
                                                 const void* gptr,
                                                 unsigned tile_d0, unsigned tile_d1,
                                                 unsigned tensor_d0, unsigned tensor_d1,
                                                 unsigned stride0) {
  const unsigned long long ga = (unsigned long long)gptr;
  u32x4 g0;
  g0[0] = 1u;                                            // count=1, user D#
  g0[1] = lds_addr;                                      // LDS byte address
  g0[2] = (unsigned)(ga & 0xFFFFFFFFull);                // global addr lo
  g0[3] = (unsigned)((ga >> 32) & 0x01FFFFFFull) | 0x80000000u; // hi | type=2
  i32x8 g1;
  g1[0] = (int)0x00010000u;                              // data_size = 2 bytes
  g1[1] = (int)((tensor_d0 & 0xFFFFu) << 16);            // tensor_dim0 lo
  g1[2] = (int)((tensor_d0 >> 16) | ((tensor_d1 & 0xFFFFu) << 16));
  g1[3] = (int)((tensor_d1 >> 16) | ((tile_d0 & 0xFFFFu) << 16));
  g1[4] = (int)(tile_d1 & 0xFFFFu);                      // tile_dim1 (tile_dim2=0)
  g1[5] = (int)stride0;                                  // tensor_dim0_stride lo
  g1[6] = 0;
  g1[7] = 0;
  i32x4 z = {0, 0, 0, 0};
#if __has_include(<hip/amd_detail/amd_gfx1250_TDM.h>)
  i32x8 z8 = {0, 0, 0, 0, 0, 0, 0, 0};
  __builtin_amdgcn_tensor_load_to_lds(g0, g1, z, z, z8, 0);   // clang-23 form
#else
  __builtin_amdgcn_tensor_load_to_lds(g0, g1, z, z, 0);       // ROCm 7.2 form
#endif
}
#endif  // HAVE_TDM

// ---------------------------------------------------------------------------
// Kernel 2: flash attention. 256 threads = 8 waves; each wave owns 16 q rows.
// K tile (32x256) and V^T tile (256x32) staged in LDS, double-buffered via
// the Tensor Data Mover so the DMA of tile j+1 overlaps the WMMAs of tile j.
// Per 32-key step: 16 WMMAs for scores + online softmax + 16 WMMAs for P*V.
// ---------------------------------------------------------------------------
__global__ void __launch_bounds__(256)
flash_attn_kernel(const _Float16* __restrict__ Qh,
                  const _Float16* __restrict__ Kh,
                  const _Float16* __restrict__ VhT,
                  float* __restrict__ out) {
  __shared__ __align__(16) _Float16 Kt[2][32][256];   // 2 x 16 KB
  __shared__ __align__(16) _Float16 Vt[2][256][32];   // 2 x 16 KB (V^T: [d][key])
  __shared__ __align__(16) _Float16 Pt[8][16][32];    // 8 KB per-wave P scratch

  const int tid  = threadIdx.x;
  const int lane = tid & 31;
  const int w    = tid >> 5;
  const int hi   = lane >> 4;
  const int nl   = lane & 15;
  const int b    = blockIdx.y;
  const int qBase = blockIdx.x * 128 + w * 16;
  const size_t qRow = (size_t)b * S + qBase;

  // Q fragments for this wave's 16 rows (already scaled by 1/sqrt(D))
  v16h qf[8];
  {
    const _Float16* qp = Qh + (qRow + nl) * D;
#pragma unroll
    for (int kc = 0; kc < 8; ++kc) {
      AFrag a;
      const _Float16* p = qp + kc * 32 + hi * 8;
      a.q[0] = *(const u32x4*)(p);
      a.q[1] = *(const u32x4*)(p + 16);
      qf[kc] = a.h;
    }
  }

  v8f acc[16];
#pragma unroll
  for (int c = 0; c < 16; ++c) acc[c] = (v8f){};
  float mrow[8], lrow[8];
#pragma unroll
  for (int r = 0; r < 8; ++r) { mrow[r] = -1e30f; lrow[r] = 0.0f; }

#if HAVE_TDM
  // Issue DMA for tile 0 from wave 0 (uniform branch; TDM issues once/wave).
  if (tid < 32) {
    tdm_load_tile_2d(lds_offset(&Kt[0][0][0]),
                     Kh + ((size_t)b * S) * D,
                     /*tile*/ D, 32, /*tensor*/ D, S, /*stride*/ D);
    tdm_load_tile_2d(lds_offset(&Vt[0][0][0]),
                     VhT + ((size_t)b * D) * S,
                     /*tile*/ 32, D, /*tensor*/ S, D, /*stride*/ S);
  }
#endif

  for (int j = 0; j < NJ; ++j) {
    const int buf = j & 1;

#if HAVE_TDM
    if (tid < 32) __builtin_amdgcn_s_wait_tensorcnt(0);
    __syncthreads();                       // tile j visible to all 8 waves
    if (tid < 32 && (j + 1) < NJ) {        // overlap DMA of tile j+1 with compute
      const int nb = (j + 1) & 1, kn = (j + 1) * 32;
      tdm_load_tile_2d(lds_offset(&Kt[nb][0][0]),
                       Kh + ((size_t)b * S + kn) * D, D, 32, D, S, D);
      tdm_load_tile_2d(lds_offset(&Vt[nb][0][0]),
                       VhT + ((size_t)b * D) * S + kn, 32, D, S, D, S);
    }
#else
    {   // fallback: cooperative copy by all 256 threads
      const int kBase = j * 32;
      const int key = tid >> 3, seg = tid & 7;
      const u32x4* ks = (const u32x4*)(Kh + ((size_t)b * S + kBase + key) * D) + seg * 4;
      u32x4* kd = (u32x4*)(&Kt[buf][key][0]) + seg * 4;
#pragma unroll
      for (int i = 0; i < 4; ++i) kd[i] = ks[i];
      const u32x4* vs = (const u32x4*)(VhT + ((size_t)b * D + tid) * S + kBase);
      u32x4* vd = (u32x4*)(&Vt[buf][tid][0]);
#pragma unroll
      for (int i = 0; i < 4; ++i) vd[i] = vs[i];
      if (j + 1 < NJ) {
        __builtin_prefetch(Kh + ((size_t)b * S + kBase + 32 + key) * D + seg * 32, 0, 1);
        __builtin_prefetch(VhT + ((size_t)b * D + tid) * S + kBase + 32, 0, 1);
      }
    }
    __syncthreads();
#endif

    const _Float16(*Ktb)[256] = Kt[buf];
    const _Float16(*Vtb)[32]  = Vt[buf];

    // ---- scores: two 16x16 tiles (keys j*32..+15 and +16..+31) ----
    v8f s0 = (v8f){}, s1 = (v8f){};
#pragma unroll
    for (int kc = 0; kc < 8; ++kc) {
      AFrag bk0, bk1;
      const _Float16* p0 = &Ktb[nl][kc * 32 + hi * 8];
      bk0.q[0] = *(const u32x4*)(p0);
      bk0.q[1] = *(const u32x4*)(p0 + 16);
      const _Float16* p1 = &Ktb[16 + nl][kc * 32 + hi * 8];
      bk1.q[0] = *(const u32x4*)(p1);
      bk1.q[1] = *(const u32x4*)(p1 + 16);
      s0 = WMMA(qf[kc], bk0.h, s0);
      s1 = WMMA(qf[kc], bk1.h, s1);
    }

    // ---- online softmax (row-wise over the 16-lane halves) ----
    float pr0[8], pr1[8], alpha[8];
#pragma unroll
    for (int r = 0; r < 8; ++r) {
      float t = fmaxf(s0[r], s1[r]);
      t = fmaxf(t, __shfl_xor(t, 1, 32));
      t = fmaxf(t, __shfl_xor(t, 2, 32));
      t = fmaxf(t, __shfl_xor(t, 4, 32));
      t = fmaxf(t, __shfl_xor(t, 8, 32));
      const float mn = fmaxf(mrow[r], t);
      alpha[r] = __expf(mrow[r] - mn);
      mrow[r]  = mn;
      pr0[r] = __expf(s0[r] - mn);
      pr1[r] = __expf(s1[r] - mn);
      float rs = pr0[r] + pr1[r];
      rs += __shfl_xor(rs, 1, 32);
      rs += __shfl_xor(rs, 2, 32);
      rs += __shfl_xor(rs, 4, 32);
      rs += __shfl_xor(rs, 8, 32);
      lrow[r] = lrow[r] * alpha[r] + rs;
    }
#pragma unroll
    for (int c = 0; c < 16; ++c)
#pragma unroll
      for (int r = 0; r < 8; ++r) acc[c][r] *= alpha[r];

    // ---- P (16x32) via per-wave LDS scratch: C-layout -> A-fragment ----
#pragma unroll
    for (int r = 0; r < 8; ++r) {
      Pt[w][r + hi * 8][nl]      = (_Float16)pr0[r];
      Pt[w][r + hi * 8][16 + nl] = (_Float16)pr1[r];
    }
    AFrag pa;
    {
      const _Float16* pp = &Pt[w][nl][hi * 8];
      pa.q[0] = *(const u32x4*)(pp);
      pa.q[1] = *(const u32x4*)(pp + 16);
    }

    // ---- P*V: 16 output chunks of 16 columns ----
#pragma unroll
    for (int c = 0; c < 16; ++c) {
      AFrag vb;
      const _Float16* vp = &Vtb[c * 16 + nl][hi * 8];
      vb.q[0] = *(const u32x4*)(vp);
      vb.q[1] = *(const u32x4*)(vp + 16);
      acc[c] = WMMA(pa.h, vb.h, acc[c]);
    }
    __syncthreads();   // all waves done with buf before DMA may overwrite it
  }

  // ---- normalize and store fp32 output (B,S,D) ----
#pragma unroll
  for (int r = 0; r < 8; ++r) {
    const float linv = 1.0f / lrow[r];
    const size_t orow = (qRow + r + hi * 8) * D;
#pragma unroll
    for (int c = 0; c < 16; ++c)
      out[orow + c * 16 + nl] = acc[c][r] * linv;
  }
}

// ---------------------------------------------------------------------------
extern "C" void kernel_launch(void* const* d_in, const int* in_sizes, int n_in,
                              void* d_out, int out_size, void* d_ws, size_t ws_size,
                              hipStream_t stream) {
  const float* x  = (const float*)d_in[0];
  const float* Wq = (const float*)d_in[1];
  const float* bq = (const float*)d_in[2];
  const float* Wk = (const float*)d_in[3];
  const float* bk = (const float*)d_in[4];
  const float* Wv = (const float*)d_in[5];
  const float* bv = (const float*)d_in[6];
  float* out = (float*)d_out;

  _Float16* Qh  = (_Float16*)d_ws;              // (B*S, D) row-major
  _Float16* Kh  = Qh + (size_t)BS * D;          // (B*S, D) row-major
  _Float16* VhT = Kh + (size_t)BS * D;          // (B, D, S)  (transposed V)

  dim3 g1(BS / 16, D / 16);
  qkv_proj_kernel<<<g1, 32, 0, stream>>>(x, Wq, bq, Qh,  0, 1.0f / 16.0f);
  qkv_proj_kernel<<<g1, 32, 0, stream>>>(x, Wk, bk, Kh,  0, 1.0f);
  qkv_proj_kernel<<<g1, 32, 0, stream>>>(x, Wv, bv, VhT, 1, 1.0f);

  dim3 g2(S / 128, Bb);
  flash_attn_kernel<<<g2, 256, 0, stream>>>(Qh, Kh, VhT, out);
}
